// FastGAT_70738111365826
// MI455X (gfx1250) — matrix operand order repaired and checked by reading, not compiled, not run on hardware
//
#include <hip/hip_runtime.h>

typedef _Float16 v16h __attribute__((ext_vector_type(16)));
typedef _Float16 v8h  __attribute__((ext_vector_type(8)));
typedef float    v8f  __attribute__((ext_vector_type(8)));

#define LRELU_ALPHA 0.2f
#define LOG2E 1.4426950408889634f

static __device__ __forceinline__ v8f wmma_f16(v16h a, v16h b, v8f c) {
  // D = A(16x32 f16) * B(32x16 f16) + C(16x16 f32)
  return __builtin_amdgcn_wmma_f32_16x16x32_f16(false, a, false, b, (short)0, c,
                                                false, false);
}

// ---------------- precision conversion kernels ----------------
__global__ void fgat_f32_to_f16(const float* __restrict__ src,
                                _Float16* __restrict__ dst, int n) {
  int i = blockIdx.x * blockDim.x + threadIdx.x;
  if (i < n) dst[i] = (_Float16)src[i];
}

// weights [K][ncol] f32 -> transposed [npad][K] f16, zero-padded rows
__global__ void fgat_w_to_f16_t(const float* __restrict__ src,
                                _Float16* __restrict__ dst,
                                int K, int ncol, int npad) {
  int i = blockIdx.x * blockDim.x + threadIdx.x;
  int n = npad * K;
  if (i >= n) return;
  int f = i / K, k = i - f * K;
  dst[i] = (f < ncol) ? (_Float16)src[(size_t)k * ncol + f] : (_Float16)0.f;
}

// ---------------- projection: Wh = A @ W, plus f_src/f_dst and WhT ----------
// One wave computes a 16-row x (NF*16) tile of Wh. Writes WhT f16 [NF*16][M]
// (feature-major, attention B operand) and fsrc/fdst = log2(e)*(Wh @ a[:F]),
// log2(e)*(Wh @ a[F:]) so the attention softmax can run in the exp2 domain
// (leaky_relu commutes with a positive scale). Wt: pre-transposed [NF*16][K].
template <int NF>
__global__ void fgat_gemm_proj(const _Float16* __restrict__ A, int K,
                               const _Float16* __restrict__ Wt, int wstride,
                               const float* __restrict__ avec, int astride,
                               _Float16* __restrict__ WhT, int whtstride,
                               float* __restrict__ fsrc, float* __restrict__ fdst,
                               int fstride, int M, int Freal) {
  int lane = threadIdx.x & 31;
  int wave = threadIdx.x >> 5;
  int tile = blockIdx.x * (blockDim.x >> 5) + wave;
  if (tile * 16 >= M) return;
  int h = blockIdx.y;
  int m = lane & 15, g = lane >> 4;
  int row = tile * 16 + m;
  const _Float16* Wp = Wt + (size_t)h * wstride;

  v8f acc[NF];
#pragma unroll
  for (int fb = 0; fb < NF; fb++)
#pragma unroll
    for (int i = 0; i < 8; i++) acc[fb][i] = 0.f;

  for (int kb = 0; kb < K; kb += 32) {
    // A fragment: lane holds row m, K = kb + 16g .. 16g+15
    const _Float16* ap = A + (size_t)row * K + kb + 16 * g;
    v8h lo = *(const v8h*)ap;
    v8h hi = *(const v8h*)(ap + 8);
    v16h afr;
#pragma unroll
    for (int t = 0; t < 8; t++) { afr[t] = lo[t]; afr[8 + t] = hi[t]; }
#pragma unroll
    for (int fb = 0; fb < NF; fb++) {
      // B fragment: lane = column n (= m) of feature block fb; k = kb+16g+t
      v16h bfr = *(const v16h*)(Wp + (size_t)(fb * 16 + m) * K + kb + 16 * g);
      acc[fb] = wmma_f16(afr, bfr, acc[fb]);
    }
  }

  // store transposed f16 copy: WhT[f][row]
  _Float16* wt = WhT + (size_t)h * whtstride;
#pragma unroll
  for (int fb = 0; fb < NF; fb++) {
    int f = fb * 16 + m;
#pragma unroll
    for (int r = 0; r < 8; r++)
      wt[(size_t)f * M + tile * 16 + r + 8 * g] = (_Float16)acc[fb][r];
  }

  // f_src/f_dst: rows live at M = r + 8g, feature per lane = fb*16 + m
  const float* av = avec + (size_t)h * astride;
#pragma unroll
  for (int r = 0; r < 8; r++) {
    float s1 = 0.f, s2 = 0.f;
#pragma unroll
    for (int fb = 0; fb < NF; fb++) {
      int f = fb * 16 + m;
      if (f < Freal) {
        float v = acc[fb][r];
        s1 += v * av[f];
        s2 += v * av[Freal + f];
      }
    }
#pragma unroll
    for (int off = 1; off < 16; off <<= 1) {
      s1 += __shfl_xor(s1, off, 32);
      s2 += __shfl_xor(s2, off, 32);
    }
    if (m == 0) {
      int rr = tile * 16 + r + 8 * g;
      fsrc[(size_t)h * fstride + rr] = s1 * LOG2E;   // exp2-domain scores
      fdst[(size_t)h * fstride + rr] = s2 * LOG2E;
    }
  }
}

// ---------------- flash-style GAT attention -------------------------------
// One wave owns 16 rows x NF*16 features. Single pass over all N columns with
// online softmax (exp2 domain); probability tile feeds the WMMA A operand.
// Accumulator rescale is skipped (wave-uniform ballot) when no row max moved.
template <int NF>
__global__ void fgat_attn(const unsigned char* __restrict__ adj,
                          const float* __restrict__ fsrc,
                          const float* __restrict__ fdst, int fstride,
                          const _Float16* __restrict__ WhT, int whtstride,
                          _Float16* outh, float* outf, int ostride, int ocolmul,
                          int N, int Freal) {
  int lane = threadIdx.x & 31;
  int wave = threadIdx.x >> 5;
  int tile = blockIdx.x * (blockDim.x >> 5) + wave;
  if (tile * 16 >= N) return;
  int h = blockIdx.y;
  int m = lane & 15, g = lane >> 4;
  int row = tile * 16 + m;

  float fs = fsrc[(size_t)h * fstride + row];
  const float* fd_base = fdst + (size_t)h * fstride;
  const _Float16* Wp = WhT + (size_t)h * whtstride;
  const unsigned char* arow = adj + (size_t)row * N;

  v8f acc[NF];
#pragma unroll
  for (int fb = 0; fb < NF; fb++)
#pragma unroll
    for (int i = 0; i < 8; i++) acc[fb][i] = 0.f;

  float rm = -3.0e38f, rd = 0.f;

  for (int jb = 0; jb < N; jb += 32) {
    int c0 = jb + 16 * g;
    // adjacency bytes for this lane's 16 K-slots (two groups of 8)
    unsigned long long ab0 = *(const unsigned long long*)(arow + c0);
    unsigned long long ab1 = *(const unsigned long long*)(arow + c0 + 8);
    float4 f0 = *(const float4*)(fd_base + c0);
    float4 f1 = *(const float4*)(fd_base + c0 + 4);
    float4 f2 = *(const float4*)(fd_base + c0 + 8);
    float4 f3 = *(const float4*)(fd_base + c0 + 12);
    float fdv[16] = {f0.x, f0.y, f0.z, f0.w, f1.x, f1.y, f1.z, f1.w,
                     f2.x, f2.y, f2.z, f2.w, f3.x, f3.y, f3.z, f3.w};
    float e[16];
#pragma unroll
    for (int t = 0; t < 16; t++) {
      unsigned long long ab = (t < 8) ? ab0 : ab1;
      unsigned int edge = (unsigned int)(ab >> (8 * (t & 7))) & 0xFFu;
      float x = fs + fdv[t];
      x = fmaxf(x, LRELU_ALPHA * x);                // leaky_relu
      e[t] = edge ? x : -9.0e15f;
    }
    float tmax = e[0];
#pragma unroll
    for (int t = 1; t < 16; t++) tmax = fmaxf(tmax, e[t]);
    tmax = fmaxf(tmax, __shfl_xor(tmax, 16, 32));   // both K halves of a row
    float nm = fmaxf(rm, tmax);
    bool upd = tmax > rm;                           // this row's max advanced

    float psum = 0.f;
    v16h pf;
#pragma unroll
    for (int t = 0; t < 16; t++) {
      float p = exp2f(e[t] - nm);                   // masked -> underflows to 0
      psum += p;
      pf[t] = (_Float16)p;
    }

    if (__ballot(upd) != 0ull) {                    // wave-uniform skip
      float scale = exp2f(rm - nm);                 // 1.0 for unchanged rows
      rd *= scale;
      float srow[8];
#pragma unroll
      for (int r = 0; r < 8; r++) srow[r] = __shfl(scale, r + 8 * g, 32);
#pragma unroll
      for (int fb = 0; fb < NF; fb++)
#pragma unroll
        for (int r = 0; r < 8; r++) acc[fb][r] *= srow[r];
    }
    rm = nm;
    rd += psum + __shfl_xor(psum, 16, 32);

#pragma unroll
    for (int fb = 0; fb < NF; fb++) {
      // B fragment: contiguous 16 halfs of WhT[f][jb + 16g ..]
      v16h bfr = *(const v16h*)(Wp + (size_t)(fb * 16 + m) * N + jb + 16 * g);
      acc[fb] = wmma_f16(pf, bfr, acc[fb]);
    }
  }

  float rinv = 1.f / rd;
  float rrow[8];
#pragma unroll
  for (int r = 0; r < 8; r++) rrow[r] = __shfl(rinv, r + 8 * g, 32);

  int ocol0 = h * ocolmul;
#pragma unroll
  for (int fb = 0; fb < NF; fb++) {
    int f = fb * 16 + m;
    if (f >= Freal) continue;
#pragma unroll
    for (int r = 0; r < 8; r++) {
      float v = acc[fb][r] * rrow[r];
      v = (v > 0.f) ? v : (__expf(v) - 1.f);        // ELU (both layers)
      int orow = tile * 16 + r + 8 * g;
      if (outh) outh[(size_t)orow * ostride + ocol0 + f] = (_Float16)v;
      else      outf[(size_t)orow * ostride + ocol0 + f] = v;
    }
  }
}

// ---------------- final log_softmax over 40 classes ------------------------
__global__ void fgat_log_softmax(float* __restrict__ out, int N, int C) {
  int i = blockIdx.x * blockDim.x + threadIdx.x;
  if (i >= N) return;
  float* r = out + (size_t)i * C;
  float mx = r[0];
  for (int c = 1; c < C; c++) mx = fmaxf(mx, r[c]);
  float s = 0.f;
  for (int c = 0; c < C; c++) s += __expf(r[c] - mx);
  float l = __logf(s) + mx;
  for (int c = 0; c < C; c++) r[c] = r[c] - l;
}

extern "C" void kernel_launch(void* const* d_in, const int* in_sizes, int n_in,
                              void* d_out, int out_size, void* d_ws, size_t ws_size,
                              hipStream_t stream) {
  (void)in_sizes; (void)n_in; (void)out_size; (void)ws_size;
  const int N = 8192, NFEAT = 512, NHID = 64, NHEADS = 4, NCLASS = 40, NPAD2 = 48;

  const float*         x   = (const float*)d_in[0];
  const unsigned char* adj = (const unsigned char*)d_in[1];   // jnp bool -> bytes
  const float*         W1  = (const float*)d_in[2];
  const float*         a1  = (const float*)d_in[3];
  const float*         Wo  = (const float*)d_in[4];
  const float*         ao  = (const float*)d_in[5];
  float* out = (float*)d_out;

  char* ws = (char*)d_ws;
  size_t off = 0;
  auto alloc = [&](size_t bytes) -> void* {
    void* p = ws + off;
    off += (bytes + 255) & ~(size_t)255;
    return p;
  };
  _Float16* xh    = (_Float16*)alloc((size_t)N * NFEAT * 2);           // 8 MB
  _Float16* W1t   = (_Float16*)alloc((size_t)NHEADS * NHID * NFEAT * 2);
  _Float16* Wot   = (_Float16*)alloc((size_t)NPAD2 * NHEADS * NHID * 2);
  _Float16* WhT1  = (_Float16*)alloc((size_t)NHEADS * NHID * N * 2);   // 4 MB
  float*    fsrc1 = (float*)alloc((size_t)NHEADS * N * 4);
  float*    fdst1 = (float*)alloc((size_t)NHEADS * N * 4);
  _Float16* x1h   = (_Float16*)alloc((size_t)N * NHEADS * NHID * 2);   // 4 MB
  _Float16* WhT2  = (_Float16*)alloc((size_t)NPAD2 * N * 2);
  float*    fsrc2 = (float*)alloc((size_t)N * 4);
  float*    fdst2 = (float*)alloc((size_t)N * 4);

  // precision conversions + weight transposes
  { int n = N * NFEAT;
    fgat_f32_to_f16<<<(n + 255) / 256, 256, 0, stream>>>(x, xh, n); }
  for (int h = 0; h < NHEADS; h++) {
    int n = NHID * NFEAT;
    fgat_w_to_f16_t<<<(n + 255) / 256, 256, 0, stream>>>(
        W1 + (size_t)h * NFEAT * NHID, W1t + (size_t)h * NHID * NFEAT,
        NFEAT, NHID, NHID);
  }
  { int n = NPAD2 * NHEADS * NHID;
    fgat_w_to_f16_t<<<(n + 255) / 256, 256, 0, stream>>>(
        Wo, Wot, NHEADS * NHID, NCLASS, NPAD2); }

  dim3 gridL1(N / 16 / 8, NHEADS);  // 8 waves/block, 16-row tiles
  dim3 gridL2(N / 16 / 8, 1);

  // layer-1 projection (4 heads): WhT(f16), f_src, f_dst
  fgat_gemm_proj<4><<<gridL1, 256, 0, stream>>>(xh, NFEAT, W1t, NHID * NFEAT,
      a1, 2 * NHID, WhT1, NHID * N, fsrc1, fdst1, N, N, NHID);

  // layer-1 flash attention (4 heads) -> x1h f16 [N][256] with ELU
  fgat_attn<4><<<gridL1, 256, 0, stream>>>(adj, fsrc1, fdst1, N, WhT1, NHID * N,
      x1h, nullptr, NHEADS * NHID, NHID, N, NHID);

  // layer-2 projection: x1h @ Wo (padded 40->48 cols)
  fgat_gemm_proj<3><<<gridL2, 256, 0, stream>>>(x1h, NHEADS * NHID, Wot, 0,
      ao, 0, WhT2, 0, fsrc2, fdst2, 0, N, NCLASS);

  // layer-2 flash attention -> d_out f32 [N][40] with ELU
  fgat_attn<3><<<gridL2, 256, 0, stream>>>(adj, fsrc2, fdst2, 0, WhT2, 0,
      nullptr, out, NCLASS, 0, N, NCLASS);

  // in-place log_softmax per row
  fgat_log_softmax<<<(N + 255) / 256, 256, 0, stream>>>(out, N, NCLASS);
}
